// CausalSelfAttention_56332791054966
// MI455X (gfx1250) — matrix operand order repaired and checked by reading, compiled, not run on hardware
//
#include <hip/hip_runtime.h>

// ---------------------------------------------------------------------------
// Types / helpers
// ---------------------------------------------------------------------------
typedef __attribute__((ext_vector_type(16))) __bf16 bf16x16;
typedef __attribute__((ext_vector_type(8)))  float  f32x8;

struct __attribute__((aligned(16))) U128 { unsigned int w[4]; };

union Frag {
  unsigned int   p[8];
  unsigned short u[16];
  bf16x16        v;
};

__device__ __forceinline__ unsigned short f2b(float f) {
  unsigned int t = __float_as_uint(f);
  t += 0x7fffu + ((t >> 16) & 1u);   // round-to-nearest-even
  return (unsigned short)(t >> 16);
}

__device__ __forceinline__ f32x8 zero8() {
  f32x8 z;
#pragma unroll
  for (int i = 0; i < 8; ++i) z[i] = 0.0f;
  return z;
}

__device__ __forceinline__ f32x8 wmma_bf16(const Frag& a, const Frag& b, f32x8 c) {
  return __builtin_amdgcn_wmma_f32_16x16x32_bf16(false, a.v, false, b.v,
                                                 (short)0, c, false, false);
}

// CDNA5 async global->LDS DMA (ASYNCcnt). LDS operand = 32-bit LDS byte
// address (low 32 bits of the flat pointer to a __shared__ object).
__device__ __forceinline__ void async_b128(unsigned lds, const void* gaddr) {
  asm volatile("global_load_async_to_lds_b128 %0, %1, off"
               :: "v"(lds), "v"(gaddr)
               : "memory");
}
__device__ __forceinline__ void wait_async0() {
  asm volatile("s_wait_asynccnt 0" ::: "memory");
}
__device__ __forceinline__ unsigned lds_addr(const void* p) {
  return (unsigned)(unsigned long long)(uintptr_t)p;  // flat->LDS offset
}

// Problem constants
static constexpr int Bc  = 2;      // batch
static constexpr int Cc  = 2048;   // sequence
static constexpr int Dc  = 2048;   // model dim
static constexpr int Hc  = 16;     // heads
static constexpr int HDc = 128;    // head dim
static constexpr int Gc  = 4;      // kv groups
static constexpr int KVc = Gc * HDc;  // 512
static constexpr int Mc  = Bc * Cc;   // 4096 rows

// ---------------------------------------------------------------------------
// fp32 -> bf16 conversion (4 elems / thread)
// ---------------------------------------------------------------------------
__global__ __launch_bounds__(256)
void f2bf_kernel(const float* __restrict__ in, unsigned short* __restrict__ out, int n) {
  int i = (blockIdx.x * 256 + threadIdx.x) * 4;
  if (i >= n) return;
  const float4 v = *(const float4*)(in + i);
  uint2 pk;
  pk.x = (unsigned int)f2b(v.x) | ((unsigned int)f2b(v.y) << 16);
  pk.y = (unsigned int)f2b(v.z) | ((unsigned int)f2b(v.w) << 16);
  *(uint2*)(out + i) = pk;
}

// ---------------------------------------------------------------------------
// Tiled bf16 WMMA GEMM: 128x128 block tile, 256 threads (8 waves),
// each wave computes 32x64 (2x4 tiles of 16x16), K-step 32.
// Tiles staged via async global->LDS DMA, double-buffered.
// MODE 0: fp32 out row-major [M][N]
// MODE 1: Q epilogue: RoPE, bf16 -> [b][h][c][128]
// MODE 2: K epilogue: RoPE, bf16 -> [b][g][c][128]
// MODE 3: V epilogue: bf16 transposed -> [b][g][128][c]
// ---------------------------------------------------------------------------
template <int MODE>
__global__ __launch_bounds__(256)
void gemm_bf16_128x128(const unsigned short* __restrict__ A,
                       const unsigned short* __restrict__ Bm,
                       int M, int N, int K, void* __restrict__ outp,
                       const float* __restrict__ cosT,
                       const float* __restrict__ sinT) {
  __shared__ unsigned short As[2][128 * 40];  // 128x32, pad->40 (16B rows)
  __shared__ unsigned short Bs[2][32 * 136];  // 32x128, pad->136

  const int tid  = threadIdx.x;
  const int lane = tid & 31, wave = tid >> 5;
  const int half = lane >> 4, ln = lane & 15;
  const int mBlk = blockIdx.y * 128, nBlk = blockIdx.x * 128;
  const int wm = (wave & 3) * 32, wn = (wave >> 2) * 64;

  f32x8 acc[2][4];
#pragma unroll
  for (int mt = 0; mt < 2; ++mt)
#pragma unroll
    for (int nt = 0; nt < 4; ++nt) acc[mt][nt] = zero8();

  // global->LDS DMA assignment: 16 bf16 (32B = 2 x b128) per thread per tile
  const int aRow = tid >> 1, aCol = (tid & 1) * 16;   // A tile 128x32
  const int bRow = tid >> 3, bCol = (tid & 7) * 16;   // B tile 32x128

  auto stage = [&](int k0, int buf) {
    const unsigned short* ga = A + (size_t)(mBlk + aRow) * K + (k0 + aCol);
    const unsigned la = lds_addr(&As[buf][aRow * 40 + aCol]);
    async_b128(la,      ga);
    async_b128(la + 16, ga + 8);
    const unsigned short* gb = Bm + (size_t)(k0 + bRow) * N + (nBlk + bCol);
    const unsigned lb = lds_addr(&Bs[buf][bRow * 136 + bCol]);
    async_b128(lb,      gb);
    async_b128(lb + 16, gb + 8);
  };

  stage(0, 0);
  for (int k0 = 0; k0 < K; k0 += 32) {
    const int buf = (k0 >> 5) & 1;
    wait_async0();        // this wave's DMA into As/Bs[buf] complete
    __syncthreads();      // ...and everyone else's; prior reads of buf^1 done
    if (k0 + 32 < K) stage(k0 + 32, buf ^ 1);

    Frag a[2], b[4];
#pragma unroll
    for (int mt = 0; mt < 2; ++mt) {
      // A 16x32 bf16 frag, contiguous k order used here (same as B): lane m=ln; lanes 0-15: K 0..15; 16-31: K 16..31
      const unsigned short* ap = &As[buf][(wm + mt * 16 + ln) * 40 + half * 16];
#pragma unroll
      for (int j = 0; j < 4; ++j) {
        a[mt].p[j]     = *(const unsigned int*)&ap[2 * j];
        a[mt].p[4 + j] = *(const unsigned int*)&ap[8 + 2 * j];
      }
    }
#pragma unroll
    for (int nt = 0; nt < 4; ++nt) {
      // B 32x16 bf16 frag, contiguous k order used here (same as A): lanes 0-15 hold K=0..15, lanes 16-31 hold K=16..31
      const unsigned short* bp = &Bs[buf][(half * 16) * 136 + wn + nt * 16 + ln];
#pragma unroll
      for (int i = 0; i < 16; ++i) b[nt].u[i] = bp[i * 136];
    }
#pragma unroll
    for (int mt = 0; mt < 2; ++mt)
#pragma unroll
      for (int nt = 0; nt < 4; ++nt)
        acc[mt][nt] = wmma_bf16(a[mt], b[nt], acc[mt][nt]);
  }

  // Epilogue. C/D layout: element j of acc = row (j + 8*half), col = ln.
#pragma unroll
  for (int mt = 0; mt < 2; ++mt) {
#pragma unroll
    for (int nt = 0; nt < 4; ++nt) {
#pragma unroll
      for (int j = 0; j < 8; ++j) {
        const int m = mBlk + wm + mt * 16 + j + 8 * half;
        const int n = nBlk + wn + nt * 16 + ln;
        float v = acc[mt][nt][j];
        if constexpr (MODE == 0) {
          ((float*)outp)[(size_t)m * N + n] = v;
        } else {
          if constexpr (MODE == 1 || MODE == 2) {
            // RoPE: pair partner (col n^1) lives in lane^1 (same row, same j).
            const int cpos = m & (Cc - 1);
            const int hp   = n & (HDc - 1);
            const float cc = cosT[cpos * (HDc / 2) + (hp >> 1)];
            const float ss = sinT[cpos * (HDc / 2) + (hp >> 1)];
            const float pr = __shfl_xor(v, 1, 32);
            v = (n & 1) ? fmaf(pr, ss, v * cc) : fmaf(v, cc, -pr * ss);
          }
          const unsigned short bv = f2b(v);
          const int bI = m >> 11;              // row / C
          const int cpos = m & (Cc - 1);
          unsigned short* o16 = (unsigned short*)outp;
          if constexpr (MODE == 1) {           // Q: [b][h][c][128]
            const int h = n >> 7, hd = n & 127;
            o16[((size_t)(bI * Hc + h) * Cc + cpos) * HDc + hd] = bv;
          } else if constexpr (MODE == 2) {    // K: [b][g][c][128]
            const int g = n >> 7, hd = n & 127;
            o16[((size_t)(bI * Gc + g) * Cc + cpos) * HDc + hd] = bv;
          } else {                             // V: transposed [b][g][128][c]
            const int g = n >> 7, hd = n & 127;
            o16[((size_t)(bI * Gc + g) * HDc + hd) * Cc + cpos] = bv;
          }
        }
      }
    }
  }
}

// ---------------------------------------------------------------------------
// Flash attention: 128 threads = 4 waves, 16 queries/wave (64/block),
// 32 keys per iteration. K/V tiles are shared by all 4 waves (same head),
// staged once per block into double-buffered LDS via async DMA.
// Online softmax; scale = 1/sqrt(D) = 1/sqrt(2048). grid = (C/64, B*H)
// ---------------------------------------------------------------------------
__global__ __launch_bounds__(128)
void attn_kernel(const unsigned short* __restrict__ qr,   // [b][h][c][128]
                 const unsigned short* __restrict__ kr,   // [b][g][c][128]
                 const unsigned short* __restrict__ vt,   // [b][g][128][c]
                 unsigned short* __restrict__ attnb) {    // [b][c][h*128+hd]
  __shared__ unsigned short Ks[2][32 * 128];   // 32 keys x 128 d   (8 KB each)
  __shared__ unsigned short Vs[2][128 * 32];   // 128 d   x 32 keys (8 KB each)
  __shared__ unsigned short Pl[4 * 16 * 40];   // per-wave P scratch (16x32 pad 40)

  const int tid  = threadIdx.x;
  const int lane = tid & 31, wave = tid >> 5;
  const int half = lane >> 4, ln = lane & 15;
  const int bh = blockIdx.y, b = bh >> 4, h = bh & 15, g = h >> 2;
  const int qblk  = blockIdx.x * 64;
  const int qbase = qblk + wave * 16;
  unsigned short* P = &Pl[wave * 16 * 40];

  const unsigned short* kbp = kr + (size_t)(b * Gc + g) * Cc * HDc;
  const unsigned short* vbp = vt + (size_t)(b * Gc + g) * HDc * Cc;

  // Q fragments (A-matrix, in registers for the whole K loop): 4 d-chunks
  Frag aq[4];
  {
    const unsigned short* qrow =
        qr + ((size_t)((b * Hc + h) * Cc + qbase + ln)) * HDc;
#pragma unroll
    for (int kb = 0; kb < 4; ++kb) {
      const unsigned short* qp = qrow + kb * 32 + half * 16;
#pragma unroll
      for (int j = 0; j < 4; ++j) {
        aq[kb].p[j]     = *(const unsigned int*)&qp[2 * j];
        aq[kb].p[4 + j] = *(const unsigned int*)&qp[8 + 2 * j];
      }
    }
  }

  // stage one 32-key K/V tile into LDS buffer `buf` (64B = 4 x b128 / thread)
  auto stage = [&](int kbase, int buf) {
    {
      const int row = tid >> 2, col = (tid & 3) * 32;     // K: 32 x 128
      const unsigned short* gk = kbp + (size_t)(kbase + row) * HDc + col;
      const unsigned lk = lds_addr(&Ks[buf][row * 128 + col]);
      async_b128(lk,      gk);
      async_b128(lk + 16, gk + 8);
      async_b128(lk + 32, gk + 16);
      async_b128(lk + 48, gk + 24);
    }
    {
      const int d = tid;                                   // V^T: 128 x 32
      const unsigned short* gv = vbp + (size_t)d * Cc + kbase;
      const unsigned lv = lds_addr(&Vs[buf][d * 32]);
      async_b128(lv,      gv);
      async_b128(lv + 16, gv + 8);
      async_b128(lv + 32, gv + 16);
      async_b128(lv + 48, gv + 24);
    }
  };

  f32x8 o[8];
#pragma unroll
  for (int t = 0; t < 8; ++t) o[t] = zero8();
  float mrow[8], lrow[8];
#pragma unroll
  for (int j = 0; j < 8; ++j) { mrow[j] = -3.0e38f; lrow[j] = 0.0f; }

  const float sc = 0.022097086912079608f;  // 1/sqrt(2048)
  // block-uniform key-block count (covers the last wave's causal range;
  // earlier waves see fully-masked rows -> p==0, harmless)
  const int nkb = (qblk + 95) >> 5;

  stage(0, 0);
  for (int kk = 0; kk < nkb; ++kk) {
    const int kbase = kk * 32;
    const int buf = kk & 1;
    wait_async0();
    __syncthreads();
    if (kk + 1 < nkb) stage(kbase + 32, buf ^ 1);

    // ---- scores: S = Q K^T (two 16-key subtiles, 4 d-chunks each) ----
    f32x8 s0 = zero8(), s1 = zero8();
#pragma unroll
    for (int kb = 0; kb < 4; ++kb) {
      Frag b0, b1;  // B 32x16: lane = key col; 16 contiguous d values (32B)
      const unsigned short* kp0 = &Ks[buf][(ln)      * 128 + kb * 32 + half * 16];
      *(U128*)&b0.p[0] = *(const U128*)kp0;
      *(U128*)&b0.p[4] = *(const U128*)(kp0 + 8);
      const unsigned short* kp1 = &Ks[buf][(16 + ln) * 128 + kb * 32 + half * 16];
      *(U128*)&b1.p[0] = *(const U128*)kp1;
      *(U128*)&b1.p[4] = *(const U128*)(kp1 + 8);
      s0 = wmma_bf16(aq[kb], b0, s0);
      s1 = wmma_bf16(aq[kb], b1, s1);
    }

    // ---- mask + online softmax (row = j + 8*half; 16 cols in half group) ----
#pragma unroll
    for (int j = 0; j < 8; ++j) {
      const int q = qbase + j + 8 * half;
      float x0 = s0[j] * sc; if (kbase + ln > q)      x0 = -3.0e38f;
      float x1 = s1[j] * sc; if (kbase + 16 + ln > q) x1 = -3.0e38f;
      float mx = fmaxf(x0, x1);
      mx = fmaxf(mx, __shfl_xor(mx, 1, 32));
      mx = fmaxf(mx, __shfl_xor(mx, 2, 32));
      mx = fmaxf(mx, __shfl_xor(mx, 4, 32));
      mx = fmaxf(mx, __shfl_xor(mx, 8, 32));
      const float mnew  = fmaxf(mrow[j], mx);
      const float alpha = __expf(mrow[j] - mnew);
      mrow[j] = mnew;
      const float p0 = __expf(x0 - mnew);
      const float p1 = __expf(x1 - mnew);
      float rs = p0 + p1;
      rs += __shfl_xor(rs, 1, 32);
      rs += __shfl_xor(rs, 2, 32);
      rs += __shfl_xor(rs, 4, 32);
      rs += __shfl_xor(rs, 8, 32);
      lrow[j] = lrow[j] * alpha + rs;
#pragma unroll
      for (int t = 0; t < 8; ++t) o[t][j] *= alpha;
      const int prow = j + 8 * half;       // C-layout -> LDS
      P[prow * 40 + ln]      = f2b(p0);
      P[prow * 40 + 16 + ln] = f2b(p1);
    }
    asm volatile("s_wait_dscnt 0" ::: "memory");  // wave-wide LDS visibility

    // ---- reload P as A-matrix fragment (16x32 over keys) ----
    Frag pa;
    {
      const unsigned short* pp = &P[ln * 40 + half * 16];
#pragma unroll
      for (int j = 0; j < 4; ++j) {
        pa.p[j]     = *(const unsigned int*)&pp[2 * j];
        pa.p[4 + j] = *(const unsigned int*)&pp[8 + 2 * j];
      }
    }

    // ---- O += P V : V^T tile [d][keys] in LDS, B frag = 32B contiguous ----
#pragma unroll
    for (int t = 0; t < 8; ++t) {
      Frag bv;
      const unsigned short* vp = &Vs[buf][(t * 16 + ln) * 32 + half * 16];
      *(U128*)&bv.p[0] = *(const U128*)vp;
      *(U128*)&bv.p[4] = *(const U128*)(vp + 8);
      o[t] = wmma_bf16(pa, bv, o[t]);
    }
  }

  // ---- finalize: O / l, store bf16 to attn_cat layout [b][c][h*128+hd] ----
#pragma unroll
  for (int t = 0; t < 8; ++t) {
#pragma unroll
    for (int j = 0; j < 8; ++j) {
      const float v = o[t][j] * (1.0f / lrow[j]);
      const int m = qbase + j + 8 * half;
      const int col = h * HDc + t * 16 + ln;
      attnb[((size_t)(b * Cc + m)) * Dc + col] = f2b(v);
    }
  }
}

// ---------------------------------------------------------------------------
// Launcher
// ---------------------------------------------------------------------------
extern "C" void kernel_launch(void* const* d_in, const int* in_sizes, int n_in,
                              void* d_out, int out_size, void* d_ws, size_t ws_size,
                              hipStream_t stream) {
  (void)in_sizes; (void)n_in; (void)out_size; (void)ws_size;
  const float* x  = (const float*)d_in[0];
  const float* fc = (const float*)d_in[1];
  const float* fs = (const float*)d_in[2];
  const float* Wq = (const float*)d_in[3];
  const float* Wk = (const float*)d_in[4];
  const float* Wv = (const float*)d_in[5];
  const float* Wo = (const float*)d_in[6];
  float* out = (float*)d_out;

  char* ws = (char*)d_ws;
  auto carve = [&](size_t elems) {
    unsigned short* p = (unsigned short*)ws;
    ws += elems * sizeof(unsigned short);
    return p;
  };
  unsigned short* xb    = carve((size_t)Mc * Dc);    // 16 MB
  unsigned short* wqb   = carve((size_t)Dc * Dc);    //  8 MB
  unsigned short* wkb   = carve((size_t)Dc * KVc);   //  2 MB
  unsigned short* wvb   = carve((size_t)Dc * KVc);   //  2 MB
  unsigned short* wob   = carve((size_t)Dc * Dc);    //  8 MB
  unsigned short* qr    = carve((size_t)Mc * Dc);    // 16 MB  [b][h][c][128]
  unsigned short* kr    = carve((size_t)Mc * KVc);   //  4 MB  [b][g][c][128]
  unsigned short* vtt   = carve((size_t)Mc * KVc);   //  4 MB  [b][g][128][c]
  unsigned short* attnb = carve((size_t)Mc * Dc);    // 16 MB  [b][c][D]

  auto cv = [&](const float* s, unsigned short* d, int n) {
    f2bf_kernel<<<dim3(n / 1024), dim3(256), 0, stream>>>(s, d, n);
  };
  cv(x,  xb,  Mc * Dc);
  cv(Wq, wqb, Dc * Dc);
  cv(Wk, wkb, Dc * KVc);
  cv(Wv, wvb, Dc * KVc);
  cv(Wo, wob, Dc * Dc);

  const dim3 blk(256);
  // Q = x@Wq (+RoPE), K = x@Wk (+RoPE), V = x@Wv (transposed store)
  gemm_bf16_128x128<1><<<dim3(Dc / 128,  Mc / 128), blk, 0, stream>>>(
      xb, wqb, Mc, Dc,  Dc, qr,  fc, fs);
  gemm_bf16_128x128<2><<<dim3(KVc / 128, Mc / 128), blk, 0, stream>>>(
      xb, wkb, Mc, KVc, Dc, kr,  fc, fs);
  gemm_bf16_128x128<3><<<dim3(KVc / 128, Mc / 128), blk, 0, stream>>>(
      xb, wvb, Mc, KVc, Dc, vtt, nullptr, nullptr);

  attn_kernel<<<dim3(Cc / 64, Bc * Hc), dim3(128), 0, stream>>>(qr, kr, vtt, attnb);

  // out = attn_cat @ Wo  (fp32 output)
  gemm_bf16_128x128<0><<<dim3(Dc / 128, Mc / 128), blk, 0, stream>>>(
      attnb, wob, Mc, Dc, Dc, out, nullptr, nullptr);
}
